// GemmaAttention_84902913507534
// MI455X (gfx1250) — matrix operand order repaired and checked by reading, not compiled, not run on hardware
//
#include <hip/hip_runtime.h>
#include <math.h>

#define NUM_HEADS_ 16
#define NUM_KV_    8
#define HEAD_DIM_  128
#define B_         2
#define S_         2048
#define H_         2048
#define SOFTCAP_   50.0f

typedef __bf16 bf16_t;
typedef bf16_t bf16x16 __attribute__((ext_vector_type(16)));
typedef bf16_t bf16x8  __attribute__((ext_vector_type(8)));
typedef bf16_t bf16x4  __attribute__((ext_vector_type(4)));
typedef float  f32x8   __attribute__((ext_vector_type(8)));
typedef float  f32x4   __attribute__((ext_vector_type(4)));

static __device__ __forceinline__ f32x8 wmma_bf16f32(bf16x16 a, bf16x16 b, f32x8 c) {
  // D(f32 16x16) = A(16x32 bf16) * B(32x16 bf16) + C
  return __builtin_amdgcn_wmma_f32_16x16x32_bf16(false, a, false, b, (short)0, c,
                                                 false, false);
}

// Build a v16 fragment from two contiguous 8-element (16B) chunks.
static __device__ __forceinline__ bf16x16 ld_frag_split(const bf16_t* lo,
                                                        const bf16_t* hi) {
  bf16x8 a = *(const bf16x8*)lo;
  bf16x8 b = *(const bf16x8*)hi;
  bf16x16 r;
#pragma unroll
  for (int j = 0; j < 8; ++j) { r[j] = a[j]; r[j + 8] = b[j]; }
  return r;
}

// ---- CDNA5 async copy: global -> LDS, 16 bytes per lane (ASYNCcnt tracked)
typedef __attribute__((address_space(3))) const void lds_cvoid_t;
static __device__ __forceinline__ unsigned lds_off_u32(const void* p) {
  return (unsigned)(unsigned long long)(lds_cvoid_t*)p;  // addrspace(3) offset
}
static __device__ __forceinline__ void async_ld_b128(unsigned lds, const void* g) {
  asm volatile("global_load_async_to_lds_b128 %0, %1, off"
               :: "v"(lds), "v"(g) : "memory");
}
static __device__ __forceinline__ void wait_async0() {
  asm volatile("s_wait_asynccnt 0x0" ::: "memory");
}
static __device__ __forceinline__ void wave_lds_fence() {
  // same-wave LDS ops are processed in order; this stops compiler reordering
  // and drains DScnt so cross-lane LDS data is visible without a full barrier
  asm volatile("s_wait_dscnt 0x0" ::: "memory");
}

// ---------------------------------------------------------------- conversion
__global__ void cvt_f32_bf16_x4(const float* __restrict__ in,
                                bf16_t* __restrict__ out, int n4) {
  int i = blockIdx.x * blockDim.x + threadIdx.x;
  if (i < n4) {
    f32x4 v = *(const f32x4*)(in + 4 * (size_t)i);
    bf16x4 o;
#pragma unroll
    for (int j = 0; j < 4; ++j) o[j] = (bf16_t)v[j];
    *(bf16x4*)(out + 4 * (size_t)i) = o;
  }
}

// ---------------------------------------------------------------- WMMA GEMM
// C[M,N] = A[M,K] * B[K,N], row-major, bf16 in, bf16/f32 out.
// Block: 256 threads (8 waves), tile 128(M) x 128(N), K stepped by 32,
// double-buffered LDS; wave computes 32x64 (8 wmma per K-step).
template <bool F32OUT>
__global__ __launch_bounds__(256, 1) void gemm_bf16(
    const bf16_t* __restrict__ A, const bf16_t* __restrict__ Bm,
    void* __restrict__ Cm, int M, int N, int K) {
  __shared__ __align__(32) bf16_t As[2][128 * 32];   // [m][k]
  __shared__ __align__(32) bf16_t BsT[2][128 * 32];  // [n][k] (transposed)

  const int tid = threadIdx.x;
  const int wave = tid >> 5, lane = tid & 31;
  const int row16 = lane & 15, hi = lane >> 4;
  const int bm = blockIdx.y * 128, bn = blockIdx.x * 128;
  const int wm = (wave & 3) * 32;
  const int wn = (wave >> 2) * 64;
  const int ar = tid >> 1, ac = (tid & 1) * 16;  // A coop: 32B/thread
  const int br = tid >> 3, bc = (tid & 7) * 16;  // B coop: 32B/thread

  auto stage = [&](int k0, int buf) {
    const bf16_t* asrc = A + (size_t)(bm + ar) * K + k0 + ac;
    async_ld_b128(lds_off_u32(&As[buf][ar * 32 + ac]), asrc);
    async_ld_b128(lds_off_u32(&As[buf][ar * 32 + ac + 8]), asrc + 8);
    const bf16_t* bsrc = Bm + (size_t)(k0 + br) * N + bn + bc;
    bf16x8 v0 = *(const bf16x8*)bsrc;
    bf16x8 v1 = *(const bf16x8*)(bsrc + 8);
#pragma unroll
    for (int j = 0; j < 8; ++j) {
      BsT[buf][(bc + j) * 32 + br]     = v0[j];
      BsT[buf][(bc + 8 + j) * 32 + br] = v1[j];
    }
  };

  f32x8 acc[2][4];
#pragma unroll
  for (int i = 0; i < 2; ++i)
#pragma unroll
    for (int j = 0; j < 4; ++j)
      acc[i][j] = (f32x8){0.f, 0.f, 0.f, 0.f, 0.f, 0.f, 0.f, 0.f};

  stage(0, 0);
  wait_async0();
  __syncthreads();

  for (int k0 = 0; k0 < K; k0 += 32) {
    const int buf = (k0 >> 5) & 1;
    if (k0 + 32 < K) stage(k0 + 32, buf ^ 1);  // prefetch next tile

    bf16x16 af[2], bf[4];
#pragma unroll
    for (int i = 0; i < 2; ++i) {
      const bf16_t* ap = &As[buf][(wm + i * 16 + row16) * 32 + hi * 8];
      af[i] = ld_frag_split(ap, ap + 16);
    }
#pragma unroll
    for (int j = 0; j < 4; ++j) {
      const bf16_t* bp = &BsT[buf][(wn + j * 16 + row16) * 32 + hi * 8];
      bf[j] = ld_frag_split(bp, bp + 16);
    }
#pragma unroll
    for (int i = 0; i < 2; ++i)
#pragma unroll
      for (int j = 0; j < 4; ++j)
        acc[i][j] = wmma_bf16f32(af[i], bf[j], acc[i][j]);

    wait_async0();
    __syncthreads();
  }

  // C layout: VGPR r -> row = r + 8*hi, col = row16
#pragma unroll
  for (int i = 0; i < 2; ++i)
#pragma unroll
    for (int j = 0; j < 4; ++j)
#pragma unroll
      for (int r = 0; r < 8; ++r) {
        size_t grow = (size_t)(bm + wm + i * 16 + r + hi * 8);
        size_t gcol = (size_t)(bn + wn + j * 16 + row16);
        if (F32OUT) ((float*)Cm)[grow * N + gcol]  = acc[i][j][r];
        else        ((bf16_t*)Cm)[grow * N + gcol] = (bf16_t)acc[i][j][r];
      }
}

// ---------------------------------------------------------------- RoPE (in place)
__global__ void rope_kernel(bf16_t* __restrict__ x, int nheads, int total) {
  int idx = blockIdx.x * blockDim.x + threadIdx.x;
  if (idx >= total) return;
  int d = idx & 63;
  int rest = idx >> 6;  // (b*S + s)*nheads + h
  int s = (rest / nheads) % S_;
  size_t base = (size_t)rest * HEAD_DIM_;
  float inv = powf(10000.0f, -((float)(2 * d)) / 128.0f);
  float f = (float)s * inv;
  float c = cosf(f), sn = sinf(f);
  float x1 = (float)x[base + d], x2 = (float)x[base + d + 64];
  x[base + d]      = (bf16_t)(x1 * c - x2 * sn);
  x[base + d + 64] = (bf16_t)(x2 * c + x1 * sn);
}

// ---------------------------------------------------------------- attention
// Q:[B,S,16,128] K,V:[B,S,8,128] bf16 (RoPE applied). Out:[B,S,2048] bf16.
__global__ __launch_bounds__(256, 1) void attn_kernel(
    const bf16_t* __restrict__ Q, const bf16_t* __restrict__ Kt,
    const bf16_t* __restrict__ Vt, bf16_t* __restrict__ Out) {
  __shared__ __align__(32) bf16_t Ks[2][32 * 128];   // [key][d]
  __shared__ __align__(32) bf16_t VsT[2][128 * 32];  // [d][key]
  __shared__ __align__(32) bf16_t Ps[8][16 * 32];    // per-wave P [row][key]

  const int tid = threadIdx.x;
  const int wave = tid >> 5, lane = tid & 31;
  const int row16 = lane & 15, hi = lane >> 4;

  const int bh = blockIdx.y;
  const int b = bh / NUM_HEADS_, h = bh % NUM_HEADS_;
  const int kvh = h >> 1;  // GQA groups = 2 (repeat_interleave)
  const int q0 = blockIdx.x * 128 + wave * 16;
  const int lr = tid >> 3, lc = (tid & 7) * 16;  // KV coop: 32B/thread each

  auto stage_kv = [&](int kb, int buf) {
    const bf16_t* ksrc =
        Kt + (((size_t)(b * S_ + kb + lr) * NUM_KV_) + kvh) * HEAD_DIM_ + lc;
    async_ld_b128(lds_off_u32(&Ks[buf][lr * 128 + lc]), ksrc);
    async_ld_b128(lds_off_u32(&Ks[buf][lr * 128 + lc + 8]), ksrc + 8);
    const bf16_t* vsrc =
        Vt + (((size_t)(b * S_ + kb + lr) * NUM_KV_) + kvh) * HEAD_DIM_ + lc;
    bf16x8 v0 = *(const bf16x8*)vsrc;
    bf16x8 v1 = *(const bf16x8*)(vsrc + 8);
#pragma unroll
    for (int j = 0; j < 8; ++j) {
      VsT[buf][(lc + j) * 32 + lr]     = v0[j];
      VsT[buf][(lc + 8 + j) * 32 + lr] = v1[j];
    }
  };

  bf16x16 qf[4];
  {
    const bf16_t* qrow =
        Q + (((size_t)(b * S_ + q0 + row16) * NUM_HEADS_) + h) * HEAD_DIM_;
#pragma unroll
    for (int ks = 0; ks < 4; ++ks) {
      const bf16_t* p = qrow + ks * 32 + hi * 8;
      qf[ks] = ld_frag_split(p, p + 16);
    }
  }

  f32x8 o[8];
  float m[8], l[8];
#pragma unroll
  for (int t = 0; t < 8; ++t) o[t] = (f32x8){0.f,0.f,0.f,0.f,0.f,0.f,0.f,0.f};
#pragma unroll
  for (int r = 0; r < 8; ++r) { m[r] = -1e30f; l[r] = 0.f; }

  const float scale = 0.08838834764831845f;  // 1/sqrt(128)

  stage_kv(0, 0);
  wait_async0();
  __syncthreads();

  for (int kb = 0; kb < S_; kb += 32) {
    const int buf = (kb >> 5) & 1;
    if (kb + 32 < S_) stage_kv(kb + 32, buf ^ 1);  // prefetch next KV tile

    // scores: 16 rows x 32 keys = two C fragments
    f32x8 c0 = (f32x8){0.f,0.f,0.f,0.f,0.f,0.f,0.f,0.f};
    f32x8 c1 = c0;
#pragma unroll
    for (int ks = 0; ks < 4; ++ks) {
      const bf16_t* p0 = &Ks[buf][row16 * 128 + ks * 32 + hi * 8];
      c0 = wmma_bf16f32(qf[ks], ld_frag_split(p0, p0 + 16), c0);
      const bf16_t* p1 = &Ks[buf][(16 + row16) * 128 + ks * 32 + hi * 8];
      c1 = wmma_bf16f32(qf[ks], ld_frag_split(p1, p1 + 16), c1);
    }

    // softcap + online softmax (rows live across each 16-lane half)
#pragma unroll
    for (int r = 0; r < 8; ++r) {
      // cap*tanh(x/cap) with tanh(y) = (e^2y - 1)/(e^2y + 1)
      float e0 = __expf(c0[r] * (2.0f * scale / SOFTCAP_));
      float e1 = __expf(c1[r] * (2.0f * scale / SOFTCAP_));
      float s0 = SOFTCAP_ * ((e0 - 1.f) / (e0 + 1.f));
      float s1 = SOFTCAP_ * ((e1 - 1.f) / (e1 + 1.f));
      float t = fmaxf(s0, s1);
#pragma unroll
      for (int off = 8; off >= 1; off >>= 1)
        t = fmaxf(t, __shfl_xor(t, off, 16));
      float mn = fmaxf(m[r], t);
      float corr = __expf(m[r] - mn);
      float p0 = __expf(s0 - mn), p1 = __expf(s1 - mn);
      c0[r] = p0; c1[r] = p1;
      float rs = p0 + p1;
#pragma unroll
      for (int off = 8; off >= 1; off >>= 1) rs += __shfl_xor(rs, off, 16);
      l[r] = l[r] * corr + rs;
      m[r] = mn;
#pragma unroll
      for (int t8 = 0; t8 < 8; ++t8) o[t8][r] *= corr;
    }

    // C-layout -> A-layout for P via wave-private LDS staging (no block barrier)
    bf16_t* ps = Ps[wave];
#pragma unroll
    for (int r = 0; r < 8; ++r) {
      ps[(r + 8 * hi) * 32 + row16]      = (bf16_t)c0[r];
      ps[(r + 8 * hi) * 32 + 16 + row16] = (bf16_t)c1[r];
    }
    wave_lds_fence();

    const bf16_t* pp = &ps[row16 * 32 + hi * 8];
    bf16x16 pf = ld_frag_split(pp, pp + 16);

    // O += P(16x32) * V(32x128)
#pragma unroll
    for (int t8 = 0; t8 < 8; ++t8) {
      const bf16_t* vp = &VsT[buf][(t8 * 16 + row16) * 32 + hi * 8];
      o[t8] = wmma_bf16f32(pf, ld_frag_split(vp, vp + 16), o[t8]);
    }

    wait_async0();
    __syncthreads();
  }

#pragma unroll
  for (int t8 = 0; t8 < 8; ++t8) {
#pragma unroll
    for (int r = 0; r < 8; ++r) {
      size_t srow = (size_t)(q0 + r + 8 * hi);
      size_t col = (size_t)(h * HEAD_DIM_ + t8 * 16 + row16);
      Out[(size_t)(b * S_ + srow) * (NUM_HEADS_ * HEAD_DIM_) + col] =
          (bf16_t)(o[t8][r] / l[r]);
    }
  }
}

// ---------------------------------------------------------------- launch
extern "C" void kernel_launch(void* const* d_in, const int* in_sizes, int n_in,
                              void* d_out, int out_size, void* d_ws,
                              size_t ws_size, hipStream_t stream) {
  const float* hs = (const float*)d_in[0];
  const float* Wq = (const float*)d_in[1];
  const float* Wk = (const float*)d_in[2];
  const float* Wv = (const float*)d_in[3];
  const float* Wo = (const float*)d_in[4];

  const size_t nX  = (size_t)B_ * S_ * H_;
  const size_t nWq = (size_t)H_ * NUM_HEADS_ * HEAD_DIM_;
  const size_t nWk = (size_t)H_ * NUM_KV_ * HEAD_DIM_;
  const size_t nQ  = (size_t)B_ * S_ * NUM_HEADS_ * HEAD_DIM_;
  const size_t nK  = (size_t)B_ * S_ * NUM_KV_ * HEAD_DIM_;

  bf16_t* Xb  = (bf16_t*)d_ws;
  bf16_t* Wqb = Xb + nX;
  bf16_t* Wkb = Wqb + nWq;
  bf16_t* Wvb = Wkb + nWk;
  bf16_t* Wob = Wvb + nWk;
  bf16_t* Qb  = Wob + nWq;
  bf16_t* Kb  = Qb + nQ;
  bf16_t* Vb  = Kb + nK;
  bf16_t* Ab  = Vb + nK;

  const int T = 256;
  cvt_f32_bf16_x4<<<(int)((nX/4  + T - 1) / T), T, 0, stream>>>(hs, Xb, (int)(nX/4));
  cvt_f32_bf16_x4<<<(int)((nWq/4 + T - 1) / T), T, 0, stream>>>(Wq, Wqb, (int)(nWq/4));
  cvt_f32_bf16_x4<<<(int)((nWk/4 + T - 1) / T), T, 0, stream>>>(Wk, Wkb, (int)(nWk/4));
  cvt_f32_bf16_x4<<<(int)((nWk/4 + T - 1) / T), T, 0, stream>>>(Wv, Wvb, (int)(nWk/4));
  cvt_f32_bf16_x4<<<(int)((nWq/4 + T - 1) / T), T, 0, stream>>>(Wo, Wob, (int)(nWq/4));

  const int M = B_ * S_;  // 4096
  gemm_bf16<false><<<dim3(2048 / 128, M / 128), T, 0, stream>>>(Xb, Wqb, Qb, M, 2048, H_);
  gemm_bf16<false><<<dim3(1024 / 128, M / 128), T, 0, stream>>>(Xb, Wkb, Kb, M, 1024, H_);
  gemm_bf16<false><<<dim3(1024 / 128, M / 128), T, 0, stream>>>(Xb, Wvb, Vb, M, 1024, H_);

  int totQ = B_ * S_ * NUM_HEADS_ * 64;
  int totK = B_ * S_ * NUM_KV_ * 64;
  rope_kernel<<<(totQ + T - 1) / T, T, 0, stream>>>(Qb, NUM_HEADS_, totQ);
  rope_kernel<<<(totK + T - 1) / T, T, 0, stream>>>(Kb, NUM_KV_, totK);

  attn_kernel<<<dim3(S_ / 128, B_ * NUM_HEADS_), T, 0, stream>>>(Qb, Kb, Vb, Ab);

  gemm_bf16<true><<<dim3(2048 / 128, M / 128), T, 0, stream>>>(Ab, Wob, d_out, M, 2048, H_);
}
